// MiddleLayer_68710886802317
// MI455X (gfx1250) — matrix operand order repaired and checked by reading, not compiled, not run on hardware
//
#include <hip/hip_runtime.h>

typedef __attribute__((ext_vector_type(16))) __bf16 v16bf;
typedef __attribute__((ext_vector_type(8)))  __bf16 v8bf;
typedef __attribute__((ext_vector_type(8)))  float  v8f;

namespace {
constexpr int EN_MID = 4096;
constexpr int K1     = 2048;   // reduction dim (odd lanes of EN_MID)
constexpr int NHALF  = 4096;   // DE_MID/2 : output pair columns
constexpr int DE     = 8192;
constexpr int BATCH  = 8192;

constexpr int M_BLK  = 128;    // rows per workgroup
constexpr int NPAIR  = 128;    // output-pair cols per workgroup (=> 256 GEMM cols)
constexpr int NCOLS  = 256;    // GEMM cols per block (left 128 = d, right 128 = d+4096)
constexpr int K_STEP = 32;     // bf16 WMMA K per step
constexpr int LROW   = 40;     // bf16 elems per LDS row: 80B stride, 16B-aligned, bank-conflict-free
constexpr int KSTEPS = K1 / K_STEP;  // 64
}

__global__ __launch_bounds__(256, 1)
void middle_layer_fused(const float* __restrict__ in_data,
                        const float* __restrict__ W,
                        const float* __restrict__ bias,
                        float* __restrict__ out)
{
    // double-buffered bf16 staging tiles: A 2*128*80B = 20KB, B 2*256*80B = 40KB (of 320KB WGP LDS)
    __shared__ __bf16 As[2][M_BLK * LROW];
    __shared__ __bf16 Bs[2][NCOLS * LROW];

    const int tid   = threadIdx.x;
    const int lane  = tid & 31;
    const int wave  = tid >> 5;          // 0..7
    const int wm    = wave >> 2;         // 0..1  : 64-row slab
    const int wn    = wave & 3;          // 0..3  : 16-col slab within each 64-col group
    const int nc    = lane & 15;
    const int hiSel = (lane >> 4) & 1;   // lane half selects K sub-range per ISA layout

    const int row0 = blockIdx.y * M_BLK;
    const int dL0  = blockIdx.x * NPAIR;

    // per-lane output pair columns and biases (two pair-groups per wave)
    const int   dA = dL0 + wn * 16 + nc;        // from GEMM cols wn*16   (L) / 128+wn*16 (R)
    const int   dB = dL0 + 64 + wn * 16 + nc;   // from GEMM cols 64+wn*16(L) / 192+wn*16 (R)
    const float bAL = bias[dA];
    const float bAR = bias[dA + NHALF];
    const float bBL = bias[dB];
    const float bBR = bias[dB + NHALF];

    // ---- staging assignments ----
    // A: thread -> (row 0..127, k-half 0/1), 16 odd elements via 8 float4
    const int aRow  = tid >> 1;
    const int aHalf = tid & 1;
    const float* aSrc = in_data + (size_t)(row0 + aRow) * EN_MID + aHalf * 32;
    // B: thread -> one full B row (32 k) of the 256-row tile
    const int bRow = tid;
    const int dRow = (bRow < NPAIR) ? (dL0 + bRow) : (NHALF + dL0 + (bRow - NPAIR));
    const float* bSrc = W + (size_t)dRow * K1;

    auto stage = [&](int buf, int k0) {
        // ---- A: 8x float4, keep .y/.w (odd elements), f32 -> bf16 ----
        const float4* ap = (const float4*)(aSrc + 2 * k0);
        v8bf a0, a1;
#pragma unroll
        for (int q = 0; q < 4; ++q) {
            float4 v = ap[q];
            a0[2 * q]     = (__bf16)v.y;
            a0[2 * q + 1] = (__bf16)v.w;
        }
#pragma unroll
        for (int q = 0; q < 4; ++q) {
            float4 v = ap[4 + q];
            a1[2 * q]     = (__bf16)v.y;
            a1[2 * q + 1] = (__bf16)v.w;
        }
        v8bf* ad = (v8bf*)&As[buf][aRow * LROW + aHalf * 16];
        ad[0] = a0;
        ad[1] = a1;

        // ---- B: one W row, 32 contiguous k via 8 float4 ----
        const float4* bp = (const float4*)(bSrc + k0);
        v8bf b01[4];
#pragma unroll
        for (int h = 0; h < 4; ++h) {
#pragma unroll
            for (int q = 0; q < 2; ++q) {
                float4 v = bp[2 * h + q];
                b01[h][4 * q]     = (__bf16)v.x;
                b01[h][4 * q + 1] = (__bf16)v.y;
                b01[h][4 * q + 2] = (__bf16)v.z;
                b01[h][4 * q + 3] = (__bf16)v.w;
            }
        }
        v8bf* bd = (v8bf*)&Bs[buf][bRow * LROW];
        bd[0] = b01[0];
        bd[1] = b01[1];
        bd[2] = b01[2];
        bd[3] = b01[3];
    };

    // A 16x32 layout (matches B): lanes 0-15 hold K0..15, lanes 16-31 hold K16..31
    auto loadA = [&](int buf, int mt) -> v16bf {
        const __bf16* p = &As[buf][(wm * 64 + mt * 16 + nc) * LROW];
        v8bf lo = *(const v8bf*)(p + hiSel * 16);       // bytes hiSel*32
        v8bf hi = *(const v8bf*)(p + hiSel * 16 + 8);   // bytes hiSel*32 + 16
        return __builtin_shufflevector(lo, hi, 0, 1, 2, 3, 4, 5, 6, 7,
                                                8, 9, 10, 11, 12, 13, 14, 15);
    };
    // 16-bit B 32x16 layout (contiguous k, same as A): lanes 0-15 hold K0..15, lanes 16-31 hold K16..31, col = lane%16
    auto loadB = [&](int buf, int nt) -> v16bf {
        const __bf16* p = &Bs[buf][(nt * 64 + wn * 16 + nc) * LROW + hiSel * 16];
        v8bf lo = *(const v8bf*)(p);
        v8bf hi = *(const v8bf*)(p + 8);
        return __builtin_shufflevector(lo, hi, 0, 1, 2, 3, 4, 5, 6, 7,
                                                8, 9, 10, 11, 12, 13, 14, 15);
    };

    v8f acc[4][4] = {};   // [mt][nt]; pairs for the fold: (nt0,nt2), (nt1,nt3)

    stage(0, 0);
    __syncthreads();

    for (int ks = 0; ks < KSTEPS; ++ks) {
        const int buf = ks & 1;
        if (ks + 1 < KSTEPS) stage(buf ^ 1, (ks + 1) * K_STEP);

        v16bf afr[4];
#pragma unroll
        for (int mt = 0; mt < 4; ++mt) afr[mt] = loadA(buf, mt);

#pragma unroll
        for (int nt = 0; nt < 4; ++nt) {
            v16bf bfr = loadB(buf, nt);
#pragma unroll
            for (int mt = 0; mt < 4; ++mt) {
                acc[mt][nt] = __builtin_amdgcn_wmma_f32_16x16x32_bf16(
                    false, afr[mt], false, bfr, (short)0, acc[mt][nt], false, false);
            }
        }
        __syncthreads();
    }

    // Epilogue: out_r = relu(accL+bL) - relu(accR+bR); write interleaved (val, 0)
    // D layout: VGPR r holds M = r + 8*hiSel, N = lane%16
#pragma unroll
    for (int mt = 0; mt < 4; ++mt) {
#pragma unroll
        for (int r = 0; r < 8; ++r) {
            const int row = row0 + wm * 64 + mt * 16 + r + hiSel * 8;
            float* orow = out + (size_t)row * DE;

            float lA = acc[mt][0][r] + bAL;  lA = lA > 0.0f ? lA : 0.0f;
            float rA = acc[mt][2][r] + bAR;  rA = rA > 0.0f ? rA : 0.0f;
            float2 oA; oA.x = lA - rA; oA.y = 0.0f;
            *(float2*)(orow + 2 * dA) = oA;

            float lB = acc[mt][1][r] + bBL;  lB = lB > 0.0f ? lB : 0.0f;
            float rB = acc[mt][3][r] + bBR;  rB = rB > 0.0f ? rB : 0.0f;
            float2 oB; oB.x = lB - rB; oB.y = 0.0f;
            *(float2*)(orow + 2 * dB) = oB;
        }
    }
}

extern "C" void kernel_launch(void* const* d_in, const int* in_sizes, int n_in,
                              void* d_out, int out_size, void* d_ws, size_t ws_size,
                              hipStream_t stream) {
    (void)in_sizes; (void)n_in; (void)out_size; (void)d_ws; (void)ws_size;
    const float* in_data = (const float*)d_in[0];
    const float* W       = (const float*)d_in[1];
    const float* bias    = (const float*)d_in[2];
    float* out           = (float*)d_out;

    dim3 grid(NHALF / NPAIR, BATCH / M_BLK);  // (32, 64)
    middle_layer_fused<<<grid, dim3(256), 0, stream>>>(in_data, W, bias, out);
}
